// MultiScaleRetention_54357106098866
// MI455X (gfx1250) — compile-verified, matrix operand order checked by reading, not run on hardware
//
#include <hip/hip_runtime.h>
#include <hip/hip_bf16.h>

// ---------------------------------------------------------------------------
// MultiScaleRetention forward for MI455X (gfx1250), wave32, WMMA bf16 path.
// bf16 operands are produced once by pre-pass kernels; hot-loop staging is
// GLOBAL_LOAD_ASYNC_TO_LDS_B128 (ASYNCcnt) with double-buffered LDS tiles,
// fragments are 2x ds_load_b128 (software-pipelined one tile ahead so LDS
// latency hides under the XDL pipe), GEMM shapes are template args.
// ---------------------------------------------------------------------------

#define B_  2
#define T_  2048
#define H_  8
#define DMODEL 1024
#define QKDIM  1024
#define VDIM   2048
#define DK     128     // per-head qk dim
#define DV     256     // per-head v dim
#define LSTR   40      // LDS row stride in ushorts (80B, multiple of 16B)

typedef __attribute__((ext_vector_type(16))) __bf16          v16bf;
typedef __attribute__((ext_vector_type(8)))  float           v8f;
typedef __attribute__((ext_vector_type(4)))  float           f32x4;
typedef __attribute__((ext_vector_type(4)))  unsigned int    u32x4;
typedef __attribute__((ext_vector_type(4)))  unsigned short  u16x4;

// float -> bf16 bits, round-to-nearest-even
__device__ __forceinline__ unsigned short f2bfbits(float f) {
    union { float f; unsigned u; } c{f};
    unsigned u = c.u;
    return (unsigned short)((u + 0x7FFFu + ((u >> 16) & 1u)) >> 16);
}
__device__ __forceinline__ float bf2f(unsigned short s) {
    union { unsigned u; float f; } c{(unsigned)s << 16};
    return c.f;
}
__device__ __forceinline__ void store_pk4(unsigned short* dst,
                                          float a, float b, float c, float d) {
    union { unsigned short s[4]; u16x4 v; } pk;
    pk.s[0] = f2bfbits(a); pk.s[1] = f2bfbits(b);
    pk.s[2] = f2bfbits(c); pk.s[3] = f2bfbits(d);
    *(u16x4*)dst = pk.v;
}

// ---- CDNA5 async global->LDS copy (16B per lane), tracked by ASYNCcnt ----
__device__ __forceinline__ void async_copy_b128(const unsigned short* lds_dst,
                                                const unsigned short* gsrc) {
    unsigned lds_addr = (unsigned)(uintptr_t)lds_dst;   // ISA: LDS addr = addr[31:0]
    unsigned long long gaddr = (unsigned long long)(uintptr_t)gsrc;
    asm volatile("global_load_async_to_lds_b128 %0, %1, off"
                 :: "v"(lds_addr), "v"(gaddr) : "memory");
}
__device__ __forceinline__ void wait_async() {
    asm volatile("s_wait_asynccnt 0x0" ::: "memory");
}

// Fragment load: two contiguous 16B runs at 8*half and 16+8*half.
// (ISA 7.12.2: 16-bit A/B frag, VGPR r<4 -> K=2r+8h, r>=4 -> K=2r+8+8h.)
__device__ __forceinline__ v16bf load_frag(const unsigned short* row, int half) {
    union { v16bf v; u32x4 q[2]; } f;
    f.q[0] = *(const u32x4*)(row + 8 * half);
    f.q[1] = *(const u32x4*)(row + 16 + 8 * half);
    return f.v;
}

// ---------------------------------------------------------------------------
// Pre-pass kernels (run once; bf16 operands reused everywhere)
// ---------------------------------------------------------------------------
__global__ void __launch_bounds__(256) cvt_bf16_kernel(
    const float* __restrict__ src, unsigned short* __restrict__ dst, int n4) {
    int i = blockIdx.x * blockDim.x + threadIdx.x;
    if (i >= n4) return;
    f32x4 v = *(const f32x4*)&src[(size_t)i * 4];
    store_pk4(&dst[(size_t)i * 4], v[0], v[1], v[2], v[3]);
}

__global__ void __launch_bounds__(256) qk_cvt_kernel(
    const float* __restrict__ qkv, unsigned short* __restrict__ qkbf) {
    int i = blockIdx.x * blockDim.x + threadIdx.x;     // B*T*512 groups
    if (i >= B_ * T_ * 512) return;
    int row = i >> 9, c4 = (i & 511) << 2;
    f32x4 v = *(const f32x4*)&qkv[(size_t)row * 4096 + c4];
    store_pk4(&qkbf[(size_t)row * 2048 + c4], v[0], v[1], v[2], v[3]);
}

__global__ void __launch_bounds__(256) transpose_cvt_kernel(
    const float* __restrict__ src, unsigned short* __restrict__ dst,
    int R, int C) {
    __shared__ float tile[32][33];
    int c0 = blockIdx.x * 32, r0 = blockIdx.y * 32;
    int lo = threadIdx.x & 31, hi = threadIdx.x >> 5;
    #pragma unroll
    for (int i = 0; i < 4; ++i) {
        int ri = hi + i * 8;
        tile[ri][lo] = src[(size_t)(r0 + ri) * C + c0 + lo];
    }
    __syncthreads();
    #pragma unroll
    for (int i = 0; i < 4; ++i) {
        int cj = hi + i * 8;
        dst[(size_t)(c0 + cj) * R + r0 + lo] = f2bfbits(tile[lo][cj]);
    }
}

// per-(b,h) transpose of a qkv section to [bh][c][t] bf16
__global__ void __launch_bounds__(256) qkv_transpose_cvt_kernel(
    const float* __restrict__ qkv, unsigned short* __restrict__ dst,
    int secOff, int Csec) {
    __shared__ float tile[32][33];                     // [t][c]
    int bh = blockIdx.z, b = bh >> 3, h = bh & 7;
    int t0 = blockIdx.x * 32, c0 = blockIdx.y * 32;
    int lo = threadIdx.x & 31, hi = threadIdx.x >> 5;
    #pragma unroll
    for (int i = 0; i < 4; ++i) {
        int tr = hi + i * 8;
        tile[tr][lo] = qkv[((size_t)b * T_ + t0 + tr) * 4096 + secOff + h * Csec + c0 + lo];
    }
    __syncthreads();
    #pragma unroll
    for (int i = 0; i < 4; ++i) {
        int cj = hi + i * 8;
        dst[((size_t)bh * Csec + c0 + cj) * T_ + t0 + lo] = f2bfbits(tile[lo][cj]);
    }
}

// ---------------------------------------------------------------------------
// C[M,N] = A[M,K] * B[K,N], A bf16 [M][K], Bt bf16 [N][K] (pre-transposed).
// N,K compile-time: all strides fold to shifts / immediate offsets.
// Double-buffered async staging: copy tile i+1 while WMMA consumes tile i.
// ---------------------------------------------------------------------------
#define GBK 32
template<int N, int K>
__global__ void __launch_bounds__(256) gemm_bf16_wmma(
    const unsigned short* __restrict__ A, const unsigned short* __restrict__ Bt,
    float* __restrict__ C) {

    __shared__ __align__(16) unsigned short As[2][128][LSTR];   // [m][k]
    __shared__ __align__(16) unsigned short Bs[2][128][LSTR];   // [n][k]

    const int tid  = threadIdx.x;
    const int lane = tid & 31, wv = tid >> 5;
    const int half = lane >> 4, ln = lane & 15;
    const int wm = wv >> 2, wn = wv & 3;          // 2x4 wave grid
    const int rowBase = blockIdx.y * 128;
    const int colBase = blockIdx.x * 128;

    // stage one 128x32 A tile + 128x32 Bt tile into buffer p (async, no wait)
    auto stage = [&](int p, int k0) {
        #pragma unroll
        for (int it = 0; it < 2; ++it) {
            int idx = tid + it * 256;
            int r = idx >> 2, kc8 = (idx & 3) << 3;
            async_copy_b128(&As[p][r][kc8], &A [(size_t)(rowBase + r) * K + k0 + kc8]);
            async_copy_b128(&Bs[p][r][kc8], &Bt[(size_t)(colBase + r) * K + k0 + kc8]);
        }
    };

    v8f acc[4][2] = {};

    stage(0, 0);
    wait_async();
    __syncthreads();

    int p = 0;
    #pragma unroll 1
    for (int k0 = 0; k0 < K; k0 += GBK) {
        if (k0 + GBK < K) stage(p ^ 1, k0 + GBK);   // overlap copy with compute

        v16bf af[4], bfv[2];
        #pragma unroll
        for (int mt = 0; mt < 4; ++mt)
            af[mt] = load_frag(&As[p][wm * 64 + mt * 16 + ln][0], half);
        #pragma unroll
        for (int nt = 0; nt < 2; ++nt)
            bfv[nt] = load_frag(&Bs[p][wn * 32 + nt * 16 + ln][0], half);

        #pragma unroll
        for (int mt = 0; mt < 4; ++mt)
            #pragma unroll
            for (int nt = 0; nt < 2; ++nt)
                acc[mt][nt] = __builtin_amdgcn_wmma_f32_16x16x32_bf16(
                    false, af[mt], false, bfv[nt], (short)0, acc[mt][nt], false, false);

        wait_async();      // next tile landed (had all of compute to do so)
        __syncthreads();
        p ^= 1;
    }

    // epilogue: lane holds col N=ln, VGPR v holds row M=v+8*half
    #pragma unroll
    for (int mt = 0; mt < 4; ++mt)
        #pragma unroll
        for (int nt = 0; nt < 2; ++nt) {
            int col   = colBase + wn * 32 + nt * 16 + ln;
            int rbase = rowBase + wm * 64 + mt * 16 + 8 * half;
            float* cp = C + (size_t)rbase * N + col;
            #pragma unroll
            for (int v = 0; v < 8; ++v)
                cp[(size_t)v * N] = acc[mt][nt][v];
        }
}

// ---------------------------------------------------------------------------
// xPos rotary applied in place to q (cols 0..1023) and k (cols 1024..2047).
// ---------------------------------------------------------------------------
__global__ void __launch_bounds__(256) xpos_kernel(float* __restrict__ qkv) {
    int idx = blockIdx.x * blockDim.x + threadIdx.x;
    if (idx >= B_ * T_ * 512) return;
    int p    = idx & 511;
    int rest = idx >> 9;
    int pos  = rest & (T_ - 1);
    int b    = rest >> 11;

    float fp = (float)p, fpos = (float)pos;
    float sv = (2.0f * fp + 0.4f * 1024.0f) / (1.4f * 1024.0f);
    float sc = exp2f(log2f(sv) * (fpos * (1.0f / 512.0f)));
    float inv_freq = exp2f(-(fp * (1.0f / 512.0f)) * log2f(10000.0f));
    float ang = fpos * inv_freq;
    float s = sinf(ang), c = cosf(ang);

    size_t base = ((size_t)b * T_ + pos) * 4096 + 2 * p;
    float qe = qkv[base], qo = qkv[base + 1];
    qkv[base]     = (qe * c - qo * s) * sc;
    qkv[base + 1] = (qo * c + qe * s) * sc;
    float rs = 1.0f / sc;
    float ke = qkv[base + 1024], ko = qkv[base + 1025];
    qkv[base + 1024] = (ke * c - ko * s) * rs;
    qkv[base + 1025] = (ko * c + ke * s) * rs;
}

// ---------------------------------------------------------------------------
// Retention: flash-style causal pass with xpos decay, per (b,h).
// Block = 128 query rows; wave w owns rows [w*16,w*16+16) x dv=256.
// Double-buffered async staging of K (32x128) and V^T (256x32) tiles.
// Fragment loads pipelined one tile ahead of their consuming WMMA.
// ---------------------------------------------------------------------------
__global__ void __launch_bounds__(256) retention_kernel(
    const unsigned short* __restrict__ qkbf, const unsigned short* __restrict__ vT,
    const int* __restrict__ tin, float* __restrict__ x) {

    __shared__ __align__(16) unsigned short Ks [2][32][DK + 8];   // [j][d]
    __shared__ __align__(16) unsigned short Vst[2][DV][LSTR];     // [v][j]
    __shared__ __align__(16) unsigned short Sld[8][16][LSTR];     // per-wave S
    __shared__ float tI[128];
    __shared__ float tJ[2][32];

    const int tid  = threadIdx.x, lane = tid & 31, wv = tid >> 5;
    const int half = lane >> 4, ln = lane & 15;
    const int bh = blockIdx.y, b = bh >> 3, h = bh & 7;
    const int rBase = blockIdx.x * 128;

    const float gamma = 1.0f - exp2f(-5.0f - (float)h);
    const float log2g = log2f(gamma);
    const float scale = 0.0883883476483184f;       // 1/sqrt(128)

    if (tid < 128) tI[tid] = (float)tin[b * T_ + rBase + tid];

    auto stage = [&](int p, int j0) {
        #pragma unroll
        for (int it = 0; it < 2; ++it) {
            int idx = tid + it * 256;              // 512 chunks (32x128)
            int j = idx >> 4, dc8 = (idx & 15) << 3;
            async_copy_b128(&Ks[p][j][dc8],
                &qkbf[((size_t)b * T_ + j0 + j) * 2048 + QKDIM + h * DK + dc8]);
        }
        #pragma unroll
        for (int it = 0; it < 4; ++it) {
            int idx = tid + it * 256;              // 1024 chunks (256x32)
            int d = idx >> 2, tc8 = (idx & 3) << 3;
            async_copy_b128(&Vst[p][d][tc8],
                &vT[((size_t)bh * DV + d) * T_ + j0 + tc8]);
        }
        if (tid < 32) tJ[p][tid] = (float)tin[b * T_ + j0 + tid];
    };

    // Q fragments straight from bf16 global (2x global_load_b128 each)
    const int rowA = rBase + wv * 16 + ln;
    const unsigned short* qrow = qkbf + ((size_t)b * T_ + rowA) * 2048 + h * DK;
    v16bf qa[4];
    #pragma unroll
    for (int f = 0; f < 4; ++f)
        qa[f] = load_frag(qrow + f * 32, half);

    v8f acc[16] = {};   // 16 rows x 256 v-cols, f32

    const int nJ = rBase / 32 + 4;                 // cover j <= rBase+127
    stage(0, 0);
    wait_async();
    __syncthreads();

    int p = 0;
    #pragma unroll 1
    for (int jb = 0; jb < nJ; ++jb) {
        const int j0 = jb * 32;
        if (jb + 1 < nJ) stage(p ^ 1, j0 + 32);    // overlap copy with compute

        // S = Q * K^T for two 16-col tiles (K-frag loads pipelined)
        #pragma unroll
        for (int nt = 0; nt < 2; ++nt) {
            int jr = nt * 16 + ln;                 // this lane's j column
            v8f sacc = {};
            v16bf kcur = load_frag(&Ks[p][jr][0], half);
            #pragma unroll
            for (int f = 0; f < 4; ++f) {
                v16bf knxt = kcur;
                if (f < 3) knxt = load_frag(&Ks[p][jr][(f + 1) * 32], half);
                sacc = __builtin_amdgcn_wmma_f32_16x16x32_bf16(
                    false, qa[f], false, kcur, (short)0, sacc, false, false);
                kcur = knxt;
            }
            // decay + causal mask, stash transposed into per-wave scratch
            #pragma unroll
            for (int v = 0; v < 8; ++v) {
                int iLoc = wv * 16 + v + 8 * half;
                int iG   = rBase + iLoc;
                int jG   = j0 + nt * 16 + ln;
                float val = 0.0f;
                if (jG <= iG) {
                    float d = tI[iLoc] - tJ[p][nt * 16 + ln];
                    val = sacc[v] * scale * exp2f(log2g * d);
                }
                Sld[wv][v + 8 * half][nt * 16 + ln] = f2bfbits(val);
            }
        }
        // same-wave LDS store->load is in order; S A-fragment = 2x b128
        v16bf sa = load_frag(&Sld[wv][ln][0], half);
        // O += S * V across dv=256; V-frag loads pipelined one tile ahead
        v16bf vcur = load_frag(&Vst[p][ln][0], half);
        #pragma unroll
        for (int ntile = 0; ntile < 16; ++ntile) {
            v16bf vnxt = vcur;
            if (ntile < 15) vnxt = load_frag(&Vst[p][(ntile + 1) * 16 + ln][0], half);
            acc[ntile] = __builtin_amdgcn_wmma_f32_16x16x32_bf16(
                false, sa, false, vcur, (short)0, acc[ntile], false, false);
            vcur = vnxt;
        }

        wait_async();
        __syncthreads();
        p ^= 1;
    }

    // write ret_out -> x[b, i, h*256 + col]
    #pragma unroll
    for (int ntile = 0; ntile < 16; ++ntile) {
        int col   = h * DV + ntile * 16 + ln;
        int iBase = rBase + wv * 16 + 8 * half;
        float* xp = x + ((size_t)b * T_ + iBase) * VDIM + col;
        #pragma unroll
        for (int v = 0; v < 8; ++v)
            xp[(size_t)v * VDIM] = acc[ntile][v];
    }
}

// ---------------------------------------------------------------------------
// current_kv[b,h] = K^T * diag(gamma^(t_last - t)) * V   (128 x 256)
// kT: bf16 [bh*128 + m][t];  vT: bf16 [bh*256 + d][t].
// ---------------------------------------------------------------------------
__global__ void __launch_bounds__(256) current_kv_kernel(
    const unsigned short* __restrict__ kT, const unsigned short* __restrict__ vT,
    const int* __restrict__ tin, float* __restrict__ kvout) {

    __shared__ __align__(16) unsigned short Ktt[DK][LSTR];   // [m][t], weighted
    __shared__ __align__(16) unsigned short Vt [DV][LSTR];   // [v][t]
    __shared__ float tw[32];

    const int tid  = threadIdx.x, lane = tid & 31, wv = tid >> 5;
    const int half = lane >> 4, ln = lane & 15;
    const int bh = blockIdx.x, b = bh >> 3, h = bh & 7;

    const float gamma = 1.0f - exp2f(-5.0f - (float)h);
    const float log2g = log2f(gamma);
    const float tlast = (float)tin[b * T_ + (T_ - 1)];

    v8f acc[16] = {};

    #pragma unroll 1
    for (int t0 = 0; t0 < T_; t0 += 32) {
        __syncthreads();
        if (tid < 32) tw[tid] = exp2f(log2g * (tlast - (float)tin[b * T_ + t0 + tid]));
        // V^T tile: pure async copy
        #pragma unroll
        for (int it = 0; it < 4; ++it) {
            int idx = tid + it * 256;              // 1024 chunks
            int d = idx >> 2, tc8 = (idx & 3) << 3;
            async_copy_b128(&Vt[d][tc8], &vT[((size_t)bh * DV + d) * T_ + t0 + tc8]);
        }
        wait_async();
        __syncthreads();
        // weighted K^T tile (small): b64 bf16 load, scale, repack
        #pragma unroll
        for (int it = 0; it < 4; ++it) {
            int idx = tid + it * 256;              // 1024 groups
            int m = idx & 127, tc4 = (idx >> 7) << 2;
            u16x4 kv = *(const u16x4*)&kT[((size_t)bh * DK + m) * T_ + t0 + tc4];
            store_pk4(&Ktt[m][tc4],
                      bf2f(kv[0]) * tw[tc4],     bf2f(kv[1]) * tw[tc4 + 1],
                      bf2f(kv[2]) * tw[tc4 + 2], bf2f(kv[3]) * tw[tc4 + 3]);
        }
        __syncthreads();

        v16bf af = load_frag(&Ktt[wv * 16 + ln][0], half);
        // V-frag loads pipelined one tile ahead of the consuming WMMA
        v16bf vcur = load_frag(&Vt[ln][0], half);
        #pragma unroll
        for (int ntile = 0; ntile < 16; ++ntile) {
            v16bf vnxt = vcur;
            if (ntile < 15) vnxt = load_frag(&Vt[(ntile + 1) * 16 + ln][0], half);
            acc[ntile] = __builtin_amdgcn_wmma_f32_16x16x32_bf16(
                false, af, false, vcur, (short)0, acc[ntile], false, false);
            vcur = vnxt;
        }
    }

    #pragma unroll
    for (int ntile = 0; ntile < 16; ++ntile) {
        int col   = ntile * 16 + ln;
        int mBase = wv * 16 + 8 * half;
        float* op = kvout + ((size_t)bh * DK + mBase) * DV + col;
        #pragma unroll
        for (int v = 0; v < 8; ++v)
            op[(size_t)v * DV] = acc[ntile][v];
    }
}

// ---------------------------------------------------------------------------
// GroupNorm(256 per head) on ret_out, fused with silu(gate); bf16 output.
// ---------------------------------------------------------------------------
__global__ void __launch_bounds__(256) gnorm_gate_kernel(
    const float* __restrict__ x, const float* __restrict__ gate,
    unsigned short* __restrict__ gbf) {
    const int lane = threadIdx.x & 31, wv = threadIdx.x >> 5;
    size_t base = (size_t)blockIdx.x * VDIM + (size_t)wv * DV;

    float v[8], sum = 0.f, sq = 0.f;
    #pragma unroll
    for (int e = 0; e < 8; ++e) {
        v[e] = x[base + lane + e * 32];
        sum += v[e];
        sq  += v[e] * v[e];
    }
    #pragma unroll
    for (int off = 16; off > 0; off >>= 1) {   // wave32 reduction
        sum += __shfl_xor(sum, off, 32);
        sq  += __shfl_xor(sq,  off, 32);
    }
    float mean = sum * (1.0f / 256.0f);
    float var  = sq * (1.0f / 256.0f) - mean * mean;
    float rstd = rsqrtf(var + 1e-5f);
    #pragma unroll
    for (int e = 0; e < 8; ++e) {
        float g    = gate[base + lane + e * 32];
        float silu = g / (1.0f + expf(-g));
        gbf[base + lane + e * 32] = f2bfbits(silu * (v[e] - mean) * rstd);
    }
}

// ---------------------------------------------------------------------------
extern "C" void kernel_launch(void* const* d_in, const int* in_sizes, int n_in,
                              void* d_out, int out_size, void* d_ws, size_t ws_size,
                              hipStream_t stream) {
    const float* hidden  = (const float*)d_in[0];
    const int*   t       = (const int*)d_in[1];
    const float* w_qkv   = (const float*)d_in[2];
    const float* w_gated = (const float*)d_in[3];
    const float* w_proj  = (const float*)d_in[4];

    float* out    = (float*)d_out;                              // (B,T,1024)
    float* kv_out = out + (size_t)B_ * T_ * DMODEL;             // (B,H,128,256)

    // workspace layout
    char* ws = (char*)d_ws;
    float* qkvBuf  = (float*)ws;                                 ws += (size_t)B_*T_*4096*4;  // 64MB
    float* xBuf    = (float*)ws;                                 ws += (size_t)B_*T_*VDIM*4;  // 32MB
    float* gateBuf = (float*)ws;                                 ws += (size_t)B_*T_*VDIM*4;  // 32MB
    unsigned short* hbf    = (unsigned short*)ws;                ws += (size_t)B_*T_*DMODEL*2;   // 8MB
    unsigned short* qkbf   = (unsigned short*)ws;                ws += (size_t)B_*T_*2048*2;     // 16MB
    unsigned short* kTbf   = (unsigned short*)ws;                ws += (size_t)B_*H_*DK*T_*2;    // 8MB
    unsigned short* vTbf   = (unsigned short*)ws;                ws += (size_t)B_*H_*DV*T_*2;    // 16MB
    unsigned short* wqkvT  = (unsigned short*)ws;                ws += (size_t)4096*1024*2;      // 8MB
    unsigned short* wgatT  = (unsigned short*)ws;                ws += (size_t)2048*1024*2;      // 4MB
    unsigned short* wprjT  = (unsigned short*)ws;                ws += (size_t)1024*2048*2;      // 4MB
    unsigned short* gbf    = (unsigned short*)ws;                ws += (size_t)B_*T_*VDIM*2;     // 16MB

    const int M = B_ * T_;   // 4096
    dim3 blk(256);

    // pre-pass converts / transposes (bf16 once, reused everywhere)
    cvt_bf16_kernel<<<(M * DMODEL / 4 + 255) / 256, blk, 0, stream>>>(hidden, hbf, M * DMODEL / 4);
    transpose_cvt_kernel<<<dim3(4096 / 32, 1024 / 32), blk, 0, stream>>>(w_qkv,  wqkvT, 1024, 4096);
    transpose_cvt_kernel<<<dim3(2048 / 32, 1024 / 32), blk, 0, stream>>>(w_gated, wgatT, 1024, 2048);
    transpose_cvt_kernel<<<dim3(1024 / 32, 2048 / 32), blk, 0, stream>>>(w_proj,  wprjT, 2048, 1024);

    // 1) qkv = hidden @ w_qkv  (f32 out, for xpos)
    gemm_bf16_wmma<4096, 1024><<<dim3(4096 / 128, M / 128), blk, 0, stream>>>(hbf, wqkvT, qkvBuf);
    // 2) xpos rotary on q and k (in place, f32)
    xpos_kernel<<<(B_ * T_ * 512 + 255) / 256, blk, 0, stream>>>(qkvBuf);
    // 2b) bf16 packs: q|k row-major, k^T and v^T per head
    qk_cvt_kernel<<<(B_ * T_ * 512 + 255) / 256, blk, 0, stream>>>(qkvBuf, qkbf);
    qkv_transpose_cvt_kernel<<<dim3(T_ / 32, DK / 32, B_ * H_), blk, 0, stream>>>(
        qkvBuf, kTbf, QKDIM, DK);
    qkv_transpose_cvt_kernel<<<dim3(T_ / 32, DV / 32, B_ * H_), blk, 0, stream>>>(
        qkvBuf, vTbf, 2 * QKDIM, DV);
    // 3) causal retention with decay -> xBuf
    retention_kernel<<<dim3(T_ / 128, B_ * H_), blk, 0, stream>>>(qkbf, vTbf, t, xBuf);
    // 4) current_kv -> second output
    current_kv_kernel<<<B_ * H_, blk, 0, stream>>>(kTbf, vTbf, t, kv_out);
    // 5) gate = hidden @ w_gated
    gemm_bf16_wmma<2048, 1024><<<dim3(VDIM / 128, M / 128), blk, 0, stream>>>(hbf, wgatT, gateBuf);
    // 6) gbf = bf16(silu(gate) * groupnorm(xBuf))
    gnorm_gate_kernel<<<B_ * T_, blk, 0, stream>>>(xBuf, gateBuf, gbf);
    // 7) out = gbf @ w_proj
    gemm_bf16_wmma<1024, 2048><<<dim3(DMODEL / 128, M / 128), blk, 0, stream>>>(gbf, wprjT, out);
}
